// MultiheadAttention_317827580792
// MI455X (gfx1250) — matrix operand order repaired and checked by reading, not compiled, not run on hardware
//
#include <hip/hip_runtime.h>

// Problem constants (from reference)
#define N_EMBED  64
#define N_HEADS  2
#define HEAD_DIM 32
#define Tt       2048
#define Bb       32

typedef __attribute__((ext_vector_type(16))) __bf16 bf16x16;
typedef __attribute__((ext_vector_type(8)))  __bf16 bf16x8;
typedef __attribute__((ext_vector_type(8)))  float  f32x8;

// ---- helpers -------------------------------------------------------------

// Native f32 -> bf16 (RNE) — emits v_cvt_pk_bf16_f32.
__device__ __forceinline__ __bf16 f2bf(float f) { return (__bf16)f; }

// WMMA wrapper: D = A(16x32 bf16) * B(32x16 bf16) + C(16x16 f32)
__device__ __forceinline__ f32x8 wmma_bf16(bf16x16 a, bf16x16 b, f32x8 c) {
  return __builtin_amdgcn_wmma_f32_16x16x32_bf16(
      /*neg_a=*/false, a, /*neg_b=*/false, b,
      /*c_mod=*/(short)0, c, /*reuse_a=*/false, /*reuse_b=*/false);
}

// A-fragment (16x32, 16-bit) layout per ISA 7.12.2: two contiguous
// 8-element (16B) runs at K = 8*hi and K = 16 + 8*hi, row M = lane%16.
__device__ __forceinline__ bf16x16 load_a_bf16(const __bf16* base, int ld, int lane) {
  int lo = lane & 15, hi = lane >> 4;
  const __bf16* row = base + (long)lo * ld + 8 * hi;
  bf16x8 c0 = *(const bf16x8*)row;
  bf16x8 c1 = *(const bf16x8*)(row + 16);
  bf16x16 a;
#pragma unroll
  for (int i = 0; i < 8; ++i) { a[i] = c0[i]; a[8 + i] = c1[i]; }
  return a;
}

__device__ __forceinline__ bf16x16 load_a_f32(const float* base, int ld, int lane) {
  int lo = lane & 15, hi = lane >> 4;
  const float* row = base + (long)lo * ld;
  bf16x16 a;
#pragma unroll
  for (int i = 0; i < 8; ++i) a[i] = f2bf(row[8 * hi + i]);
#pragma unroll
  for (int i = 0; i < 8; ++i) a[8 + i] = f2bf(row[16 + 8 * hi + i]);
  return a;
}

// B-fragment (32x16, 16-bit): lane holds column n = lane%16,
// two 8-element runs at K = 8*(lane/16) and K = 16 + 8*(lane/16).  Source layout: B[k][n] = base[n*ld + k],
// so each lane reads two 8-element runs (16B each) — two vectorized 16B loads.
__device__ __forceinline__ bf16x16 load_b_bf16(const __bf16* base, long ld, int lane) {
  int lo = lane & 15, hi = lane >> 4;
  const __bf16* p = base + (long)lo * ld + 8 * hi;
  bf16x8 c0 = *(const bf16x8*)p;
  bf16x8 c1 = *(const bf16x8*)(p + 16);
  bf16x16 b;
#pragma unroll
  for (int i = 0; i < 8; ++i) { b[i] = c0[i]; b[8 + i] = c1[i]; }
  return b;
}

// Async global->LDS copy: each lane moves 16B; tracked by ASYNCcnt.
__device__ __forceinline__ void async_b128(unsigned lds_byte_addr, const void* gptr) {
  asm volatile("global_load_async_to_lds_b128 %0, %1, off"
               :: "v"(lds_byte_addr), "v"((unsigned long long)(size_t)gptr)
               : "memory");
}

// ---- Pass 0: one-time weight conversion f32 -> bf16 ----------------------
// Wb layout: [Wq | Wk | Wv | Wp], each 64*64 bf16 (32 KB total).
__global__ __launch_bounds__(256)
void wconv_kernel(const float* __restrict__ Wq, const float* __restrict__ Wk,
                  const float* __restrict__ Wv, const float* __restrict__ Wp,
                  __bf16* __restrict__ Wb) {
  int i = blockIdx.x * 256 + threadIdx.x;  // 0 .. 16383
  const float* srcs[4] = {Wq, Wk, Wv, Wp};
  Wb[i] = f2bf(srcs[i >> 12][i & 4095]);
}

// ---- Pass 1: fused QKV projection (one wave per 16-token tile) -----------
// Q,K stored [b*T + t][64] bf16 ; V stored transposed [b][c][T] bf16.
__global__ __launch_bounds__(32)
void qkv_proj_kernel(const float* __restrict__ x,
                     const __bf16* __restrict__ Wb,   // bf16 weights
                     __bf16* __restrict__ Qws,
                     __bf16* __restrict__ Kws,
                     __bf16* __restrict__ Vt) {
  int lane = threadIdx.x & 31;
  int lo = lane & 15, hi = lane >> 4;
  long g = (long)blockIdx.x * 16;  // first global token row of this tile

  const float* xrow = x + g * N_EMBED;
  bf16x16 a0 = load_a_f32(xrow, N_EMBED, lane);       // c = 0..31
  bf16x16 a1 = load_a_f32(xrow + 32, N_EMBED, lane);  // c = 32..63

#pragma unroll
  for (int w = 0; w < 3; ++w) {
    const __bf16* W = Wb + (long)w * N_EMBED * N_EMBED;  // row-major [o][c]
#pragma unroll
    for (int nt = 0; nt < 4; ++nt) {
      f32x8 acc = {};
      acc = wmma_bf16(a0, load_b_bf16(W + (long)nt * 16 * N_EMBED + 0,  N_EMBED, lane), acc);
      acc = wmma_bf16(a1, load_b_bf16(W + (long)nt * 16 * N_EMBED + 32, N_EMBED, lane), acc);
      // D layout: lane col n = lo, rows m = r + 8*hi
      if (w == 2) {
        long b = g / Tt, t0 = g % Tt;
        long c = nt * 16 + lo;
        __bf16* dst = Vt + (b * N_EMBED + c) * Tt + t0 + 8 * hi;
        bf16x8 pk;
#pragma unroll
        for (int r = 0; r < 8; ++r) pk[r] = f2bf(acc[r]);
        *(bf16x8*)dst = pk;  // single 16B store, contiguous in t
      } else {
        __bf16* dst = (w == 0 ? Qws : Kws) + (g + 8 * hi) * N_EMBED + nt * 16 + lo;
#pragma unroll
        for (int r = 0; r < 8; ++r) dst[(long)r * N_EMBED] = f2bf(acc[r]);
      }
    }
  }
}

// ---- Pass 2: flash attention with async double-buffered K/V staging ------
// One wave per (b, head, 16-query tile).
// LDS map (bytes):  [0,4K)  K chunk double buffer (2 x 32x32 bf16)
//                   [4K,8K) V chunk double buffer (2 x 32x32 bf16)
//                   [8K,9K) P tile (16x32 bf16) for D->A transpose
__global__ __launch_bounds__(32)
void attention_kernel(const __bf16* __restrict__ Qws,
                      const __bf16* __restrict__ Kws,
                      const __bf16* __restrict__ Vt,
                      const int* __restrict__ mask,  // (B,1,1,T) flat
                      __bf16* __restrict__ att) {
  __shared__ __align__(1024) char smem[9216];

  int lane = threadIdx.x & 31;
  int lo = lane & 15, hi = lane >> 4;
  int qt = blockIdx.x % (Tt / 16);
  int h  = (blockIdx.x / (Tt / 16)) % N_HEADS;
  int b  =  blockIdx.x / ((Tt / 16) * N_HEADS);
  int qbase = qt * 16;
  int cOff  = h * HEAD_DIM;

  // Generic shared pointer low 32 bits == workgroup-relative LDS byte address
  // (ISA §10.2: LDS aperture uses addr[31:0]).
  unsigned smem_lds = (unsigned)(size_t)(void*)smem;

  const __bf16* Kbase = Kws + (long)b * Tt * N_EMBED + cOff;       // [t][64] slice
  const __bf16* Vbase = Vt + ((long)b * N_EMBED + cOff) * Tt;      // [d][T] slice
  __bf16* Plds = (__bf16*)(smem + 8192);

  bf16x16 qf = load_a_bf16(Qws + ((long)b * Tt + qbase) * N_EMBED + cOff, N_EMBED, lane);

  const float scale  = 0.1767766952966369f;  // 1/sqrt(32)
  const float neginf = -__builtin_inff();

  float m[8], l[8];
  f32x8 o0 = {}, o1 = {};
#pragma unroll
  for (int r = 0; r < 8; ++r) { m[r] = neginf; l[r] = 0.0f; }

  int nPairs = qt / 2 + 1;  // 32-key chunks needed under causal mask
  int row = lane >> 2, seg = lane & 3;  // 8 rows x 64B per async instruction

  // Prefetch chunk 0 into buffer 0: 4 instrs for K (32 rows x 64B), 4 for V.
  {
#pragma unroll
    for (int i = 0; i < 4; ++i)
      async_b128(smem_lds + lane * 16 + i * 512,
                 (const char*)(Kbase + (long)(row + 8 * i) * N_EMBED) + seg * 16);
#pragma unroll
    for (int i = 0; i < 4; ++i)
      async_b128(smem_lds + 4096 + lane * 16 + i * 512,
                 (const char*)(Vbase + (long)(row + 8 * i) * Tt) + seg * 16);
  }

  for (int kp = 0; kp < nPairs; ++kp) {
    int  buf  = kp & 1;
    bool more = (kp + 1) < nPairs;
    if (more) {  // prefetch next chunk into the other buffer
      int nb = buf ^ 1;
      long kt = (long)(kp + 1) * 32;
#pragma unroll
      for (int i = 0; i < 4; ++i)
        async_b128(smem_lds + nb * 2048 + lane * 16 + i * 512,
                   (const char*)(Kbase + (kt + row + 8 * i) * N_EMBED) + seg * 16);
#pragma unroll
      for (int i = 0; i < 4; ++i)
        async_b128(smem_lds + 4096 + nb * 2048 + lane * 16 + i * 512,
                   (const char*)(Vbase + (long)(row + 8 * i) * Tt + kt) + seg * 16);
      asm volatile("s_wait_asynccnt 0x8" ::: "memory");  // current chunk landed
    } else {
      asm volatile("s_wait_asynccnt 0x0" ::: "memory");
    }

    const __bf16* Kl = (const __bf16*)(smem + buf * 2048);         // [32 tok][32 d]
    const __bf16* Vl = (const __bf16*)(smem + 4096 + buf * 2048);  // [32 d][32 tok]

    f32x8 s[2];
#pragma unroll
    for (int sub = 0; sub < 2; ++sub) {
      bf16x16 kf = load_b_bf16(Kl + sub * 16 * 32, 32, lane);  // ds_load_b128 x2
      f32x8 z = {};
      z = wmma_bf16(qf, kf, z);  // S = Q K^T  (K-dim = hd = 32, one WMMA)
      int  key = kp * 32 + sub * 16 + lo;  // D layout: lane col = key index
      bool ok  = mask[(long)b * Tt + key] != 0;
#pragma unroll
      for (int r = 0; r < 8; ++r) {
        float v = z[r] * scale;
        int query = qbase + r + 8 * hi;
        if (key > query || !ok) v = neginf;
        z[r] = v;
      }
      s[sub] = z;
    }

    // Online softmax over the 32-key chunk. Row M = r + 8*hi lives entirely
    // in one 16-lane group -> 4-step shfl_xor reductions.
#pragma unroll
    for (int r = 0; r < 8; ++r) {
      float rmax = fmaxf(s[0][r], s[1][r]);
      rmax = fmaxf(rmax, __shfl_xor(rmax, 1));
      rmax = fmaxf(rmax, __shfl_xor(rmax, 2));
      rmax = fmaxf(rmax, __shfl_xor(rmax, 4));
      rmax = fmaxf(rmax, __shfl_xor(rmax, 8));
      float mn    = fmaxf(m[r], rmax);
      float alpha = __expf(m[r] - mn);
      float p0 = __expf(s[0][r] - mn);
      float p1 = __expf(s[1][r] - mn);
      float rs = p0 + p1;
      rs += __shfl_xor(rs, 1);
      rs += __shfl_xor(rs, 2);
      rs += __shfl_xor(rs, 4);
      rs += __shfl_xor(rs, 8);
      l[r] = l[r] * alpha + rs;
      m[r] = mn;
      o0[r] *= alpha;
      o1[r] *= alpha;
      // stage P (D layout) to LDS row-major [16][32]
      Plds[(r + 8 * hi) * 32 + lo]      = f2bf(p0);
      Plds[(r + 8 * hi) * 32 + 16 + lo] = f2bf(p1);
    }

    // Reload P as an A-fragment (16x32) — single wave, DS in-order.
    bf16x16 pf;
    {
      const __bf16* prow = &Plds[lo * 32 + 8 * hi];
      bf16x8 c0 = *(const bf16x8*)prow;
      bf16x8 c1 = *(const bf16x8*)(prow + 16);
#pragma unroll
      for (int i = 0; i < 8; ++i) { pf[i] = c0[i]; pf[8 + i] = c1[i]; }
    }

    // V B-fragments from LDS: B[k=key][n=d] = Vl[n][k]
    bf16x16 v0 = load_b_bf16(Vl, 32, lane);            // d = 0..15
    bf16x16 v1 = load_b_bf16(Vl + 16 * 32, 32, lane);  // d = 16..31
    o0 = wmma_bf16(pf, v0, o0);
    o1 = wmma_bf16(pf, v1, o1);
  }

  // Normalize and write attention output (bf16, [b*T + t][64])
  __bf16* dst = att + ((long)b * Tt + qbase + 8 * hi) * N_EMBED + cOff + lo;
#pragma unroll
  for (int r = 0; r < 8; ++r) {
    float rl = 1.0f / l[r];
    dst[(long)r * N_EMBED]      = f2bf(o0[r] * rl);
    dst[(long)r * N_EMBED + 16] = f2bf(o1[r] * rl);
  }
}

// ---- Pass 3: output projection + bias (f32 out) --------------------------
__global__ __launch_bounds__(32)
void out_proj_kernel(const __bf16* __restrict__ att,
                     const __bf16* __restrict__ Wpb,  // bf16 Wp
                     const float* __restrict__ bp,
                     float* __restrict__ out) {
  int lane = threadIdx.x & 31;
  int lo = lane & 15, hi = lane >> 4;
  long g = (long)blockIdx.x * 16;

  const __bf16* arow = att + g * N_EMBED;
  bf16x16 a0 = load_a_bf16(arow, N_EMBED, lane);
  bf16x16 a1 = load_a_bf16(arow + 32, N_EMBED, lane);

#pragma unroll
  for (int nt = 0; nt < 4; ++nt) {
    f32x8 acc = {};
    acc = wmma_bf16(a0, load_b_bf16(Wpb + (long)nt * 16 * N_EMBED + 0,  N_EMBED, lane), acc);
    acc = wmma_bf16(a1, load_b_bf16(Wpb + (long)nt * 16 * N_EMBED + 32, N_EMBED, lane), acc);
    float bias = bp[nt * 16 + lo];
    float* dst = out + (g + 8 * hi) * N_EMBED + nt * 16 + lo;
#pragma unroll
    for (int r = 0; r < 8; ++r) dst[(long)r * N_EMBED] = acc[r] + bias;
  }
}

// ---- launch --------------------------------------------------------------
extern "C" void kernel_launch(void* const* d_in, const int* in_sizes, int n_in,
                              void* d_out, int out_size, void* d_ws, size_t ws_size,
                              hipStream_t stream) {
  const float* idx  = (const float*)d_in[0];
  const int*   mask = (const int*)  d_in[1];
  const float* Wq   = (const float*)d_in[2];
  const float* Wk   = (const float*)d_in[3];
  const float* Wv   = (const float*)d_in[4];
  const float* Wp   = (const float*)d_in[5];
  const float* bp   = (const float*)d_in[6];
  float* out = (float*)d_out;

  // Workspace (bf16): Q, K, Vt, att planes (8 MB each) + converted weights.
  size_t plane = (size_t)Bb * Tt * N_EMBED;
  __bf16* Qws = (__bf16*)d_ws;
  __bf16* Kws = Qws + plane;
  __bf16* Vt  = Kws + plane;
  __bf16* att = Vt + plane;
  __bf16* Wb  = att + plane;  // 4 * 64*64 bf16 = 32 KB

  wconv_kernel<<<(4 * N_EMBED * N_EMBED) / 256, 256, 0, stream>>>(Wq, Wk, Wv, Wp, Wb);

  int ntiles = Bb * Tt / 16;  // 4096 token tiles
  qkv_proj_kernel<<<ntiles, 32, 0, stream>>>(idx, Wb, Qws, Kws, Vt);
  attention_kernel<<<Bb * N_HEADS * (Tt / 16), 32, 0, stream>>>(Qws, Kws, Vt, mask, att);
  out_proj_kernel<<<ntiles, 32, 0, stream>>>(att, Wb + 3 * N_EMBED * N_EMBED, bp, out);
}
